// HighwayLayerDiscrete_47313359733358
// MI455X (gfx1250) — matrix operand order repaired and checked by reading, not compiled, not run on hardware
//
#include <hip/hip_runtime.h>

#define UNITS   1024
#define EMBED   512
#define BATCH   64
#define SEQ     256
#define NLAYER  2
#define ALPHA   0.2f

#define YSTR    1040   // bf16 elems per row, y/act LDS buffers (pad vs bank conflicts)
#define FSTR    1032   // f32 elems per row, persistent y state

typedef __attribute__((ext_vector_type(16))) __bf16 v16bf;
typedef __attribute__((ext_vector_type(8)))  float  v8f;
typedef __attribute__((ext_vector_type(4)))  float  f32x4;
typedef __attribute__((ext_vector_type(4)))  __bf16 bf16x4;
typedef __attribute__((ext_vector_type(4)))  unsigned int u32x4;

union ABu { v16bf v; u32x4 u[2]; };

// ---------------- prep: f32 [K][N] -> bf16 transposed [N][K] ----------------
__global__ void tconv_kernel(const float* __restrict__ src, __bf16* __restrict__ dst,
                             int K, int N) {
    int i = blockIdx.x * blockDim.x + threadIdx.x;
    if (i < K * N) {
        int k = i / N, n = i % N;
        dst[(size_t)n * K + k] = (__bf16)src[i];
    }
}

// ---------------- prep: embedding gather -> bf16 time-major [T][B][E] -------
__global__ void gather_kernel(const int* __restrict__ x, const float* __restrict__ emb,
                              __bf16* __restrict__ xe) {
    int bt = blockIdx.x;            // b * SEQ + t
    int b = bt / SEQ, t = bt % SEQ;
    int idx = x[(size_t)b * SEQ + t];
    const float* s = emb + (size_t)idx * EMBED;
    __bf16* d = xe + ((size_t)t * BATCH + b) * EMBED;
    for (int e = threadIdx.x; e < EMBED; e += blockDim.x)
        d[e] = (__bf16)s[e];
}

// ---------------- WMMA fragment loads ----------------
// A fragment (16x32 bf16) from LDS, ISA 7.12.2 layout:
// lane half 'hl' holds K = k0 + hl*8 + {0..7} and k0 + 16 + hl*8 + {0..7}.
__device__ __forceinline__ v16bf load_a_lds(const __bf16* base, int astride,
                                            int r, int hl, int k0) {
    const __bf16* p = base + (size_t)r * astride + k0 + hl * 8;
    ABu t;
    t.u[0] = *reinterpret_cast<const u32x4*>(p);
    t.u[1] = *reinterpret_cast<const u32x4*>(p + 16);
    return t.v;
}

// Same A layout but sourced from global memory (row-major, leading dim lda).
__device__ __forceinline__ v16bf load_a_g(const __bf16* __restrict__ A, int lda,
                                          int r, int hl, int k0) {
    const __bf16* p = A + (size_t)r * lda + k0 + hl * 8;
    ABu t;
    t.u[0] = *reinterpret_cast<const u32x4*>(p);
    t.u[1] = *reinterpret_cast<const u32x4*>(p + 16);
    return t.v;
}

// B fragment (32x16 bf16) from transposed weights WT[N][K]:
// lane (hl, r): column n0+r, K = k0 + hl*8 + {0..7} and k0 + 16 + hl*8 + {0..7}.
__device__ __forceinline__ v16bf load_b_g(const __bf16* __restrict__ WT, int ldk,
                                          int n0, int r, int hl, int k0) {
    const __bf16* p = WT + (size_t)(n0 + r) * ldk + k0 + hl * 8;
    ABu t;
    t.u[0] = *reinterpret_cast<const u32x4*>(p);
    t.u[1] = *reinterpret_cast<const u32x4*>(p + 16);
    return t.v;
}

// Accumulate 4 independent 16x16 tiles over K-blocks of 32 (A in LDS).
// K-loop kept rolled: 2 ds_load_b128 + 8 global_load_b128 + 4 wmma, no spills.
__device__ __forceinline__ void gemm_acc(v8f (&c)[4],
                                         const __bf16* Alds, int astride,
                                         const __bf16* __restrict__ WT, int ldk,
                                         int kblocks, int wave, int hl, int r) {
#pragma clang loop unroll(disable)
    for (int kb = 0; kb < kblocks; ++kb) {
        int k0 = kb * 32;
        v16bf a = load_a_lds(Alds, astride, r, hl, k0);
#pragma unroll
        for (int i = 0; i < 4; ++i) {
            int n0 = (wave + 16 * i) * 16;
            v16bf b = load_b_g(WT, ldk, n0, r, hl, k0);
            c[i] = __builtin_amdgcn_wmma_f32_16x16x32_bf16(
                false, a, false, b, (short)0, c[i], false, false);
        }
    }
}

__device__ __forceinline__ void zero4(v8f (&c)[4]) {
    v8f z = {0.f, 0.f, 0.f, 0.f, 0.f, 0.f, 0.f, 0.f};
#pragma unroll
    for (int i = 0; i < 4; ++i) c[i] = z;
}

__device__ __forceinline__ float lrelu(float v) {
    return fmaxf(v, ALPHA * v);   // valid leaky-relu since ALPHA < 1
}

// ---------------- time-parallel precompute: P = xe @ Wx + b_in ----------------
// xe@Wx has no recurrence: do all T*B=16384 rows in one chip-wide WMMA GEMM,
// removing 1 MB/step of weight traffic + xe staging from the serial loop.
// Each WG: 64 rows x 256 cols; wave w -> m-tile w/4, n-tiles (w%4)*4..+3.
__global__ __launch_bounds__(512, 1) void pregemm_kernel(
    const __bf16* __restrict__ X,    // [SEQ*BATCH][EMBED] bf16, time-major
    const __bf16* __restrict__ wxT,  // [UNITS][EMBED]
    const float* __restrict__ b_in,
    float* __restrict__ P) {         // [SEQ*BATCH][UNITS] f32
    const int tid = threadIdx.x, wave = tid >> 5, lane = tid & 31;
    const int hl = lane >> 4, r = lane & 15;
    const int row0 = blockIdx.x * 64 + (wave >> 2) * 16;
    const int colb = blockIdx.y * 256 + (wave & 3) * 64;

    v8f c[4];
    zero4(c);
    const __bf16* A = X + (size_t)row0 * EMBED;
#pragma clang loop unroll(disable)
    for (int kb = 0; kb < EMBED / 32; ++kb) {
        int k0 = kb * 32;
        v16bf a = load_a_g(A, EMBED, r, hl, k0);
#pragma unroll
        for (int i = 0; i < 4; ++i) {
            v16bf b = load_b_g(wxT, EMBED, colb + i * 16, r, hl, k0);
            c[i] = __builtin_amdgcn_wmma_f32_16x16x32_bf16(
                false, a, false, b, (short)0, c[i], false, false);
        }
    }
#pragma unroll
    for (int i = 0; i < 4; ++i) {
        int col = colb + i * 16 + r;
        float bi = b_in[col];
#pragma unroll
        for (int j = 0; j < 8; ++j) {
            int row = row0 + hl * 8 + j;
            P[(size_t)row * UNITS + col] = c[i][j] + bi;
        }
    }
}

// ---------------- persistent recurrent kernel ----------------
// grid = BATCH/16 WGs; each owns 16 batch rows for all 256 steps (rows are
// independent: every GEMM mixes only features -> no inter-WG sync needed).
// 512 threads = 16 wave32; wave w owns column tiles {w, w+16, w+32, w+48}.
__global__ __launch_bounds__(512, 1) void hw_rnn_kernel(
    const float* __restrict__ b_hid, const float* __restrict__ b_out,
    const float* __restrict__ h0,
    const __bf16* __restrict__ wyT, const __bf16* __restrict__ whT,
    const __bf16* __restrict__ woT,
    const float* __restrict__ P, float* __restrict__ out) {

    extern __shared__ __align__(16) char lds_raw[];
    __bf16* ybf  = reinterpret_cast<__bf16*>(lds_raw);            // [16][YSTR]
    __bf16* act  = reinterpret_cast<__bf16*>(lds_raw + 33280);    // [16][YSTR]
    float*  yf32 = reinterpret_cast<float*>(lds_raw + 66560);     // [16][FSTR]

    const int tid  = threadIdx.x;
    const int wave = tid >> 5;
    const int lane = tid & 31;
    const int hl   = lane >> 4;
    const int r    = lane & 15;
    const int m0   = blockIdx.x * 16;       // first batch row of this WG

    int col[4];
#pragma unroll
    for (int i = 0; i < 4; ++i) col[i] = (wave + 16 * i) * 16 + r;

    // init persistent y state in LDS: broadcast h0 across the 16 batch rows
    for (int idx = tid; idx < 16 * UNITS; idx += 512) {
        int row = idx >> 10, u = idx & (UNITS - 1);
        yf32[(size_t)row * FSTR + u] = h0[u];
    }
    __syncthreads();

    for (int t = 0; t < SEQ; ++t) {
        // 1) publish y as bf16 for the K-dim of GEMM1 (f32x4 -> bf16x4)
        {
            int base = tid * 32;            // 512 thr * 32 elem = 16*1024
            int row = base >> 10, kk = base & (UNITS - 1);
            const float* s = yf32 + (size_t)row * FSTR + kk;
            __bf16* d = ybf + (size_t)row * YSTR + kk;
#pragma unroll
            for (int q = 0; q < 8; ++q) {
                f32x4 v = *reinterpret_cast<const f32x4*>(s + q * 4);
                *reinterpret_cast<bf16x4*>(d + q * 4) =
                    __builtin_convertvector(v, bf16x4);
            }
        }
        __syncthreads();

        v8f c[4];

        // 2) GEMM1: c = P[t] (xe@Wx + b_in, precomputed) + y @ Wy
#pragma unroll
        for (int i = 0; i < 4; ++i) {
#pragma unroll
            for (int j = 0; j < 8; ++j) {
                int row = hl * 8 + j;
                c[i][j] = P[((size_t)(t * BATCH + m0 + row)) * UNITS + col[i]];
            }
        }
        gemm_acc(c, ybf, YSTR, wyT, UNITS, UNITS / 32, wave, hl, r);
#pragma unroll
        for (int i = 0; i < 4; ++i)
#pragma unroll
            for (int j = 0; j < 8; ++j)
                act[(size_t)(hl * 8 + j) * YSTR + col[i]] = (__bf16)lrelu(c[i][j]);
        __syncthreads();

        // 3) hidden layers (single act buffer, two barriers per layer)
        for (int l = 0; l < NLAYER; ++l) {
            zero4(c);
            gemm_acc(c, act, YSTR, whT + (size_t)l * UNITS * UNITS, UNITS,
                     UNITS / 32, wave, hl, r);
            __syncthreads();                // all waves done READING act
#pragma unroll
            for (int i = 0; i < 4; ++i) {
                float bi = b_hid[(size_t)l * UNITS + col[i]];
#pragma unroll
                for (int j = 0; j < 8; ++j)
                    act[(size_t)(hl * 8 + j) * YSTR + col[i]] =
                        (__bf16)lrelu(c[i][j] + bi);
            }
            __syncthreads();                // act ready for next consumer
        }

        // 4) output projection + highway residual into LDS-resident y
        zero4(c);
        gemm_acc(c, act, YSTR, woT, UNITS, UNITS / 32, wave, hl, r);
#pragma unroll
        for (int i = 0; i < 4; ++i) {
            float bo = b_out[col[i]];
#pragma unroll
            for (int j = 0; j < 8; ++j) {
                int row = hl * 8 + j;
                float yv = yf32[(size_t)row * FSTR + col[i]] + c[i][j] + bo;
                yf32[(size_t)row * FSTR + col[i]] = yv;
                out[((size_t)(m0 + row) * SEQ + t) * UNITS + col[i]] = yv;
            }
        }
        __syncthreads();                    // yf32 stable before next-step copy
    }
}

// ---------------- host-side launch ----------------
extern "C" void kernel_launch(void* const* d_in, const int* in_sizes, int n_in,
                              void* d_out, int out_size, void* d_ws, size_t ws_size,
                              hipStream_t stream) {
    const int*   x    = (const int*)  d_in[0];
    const float* emb  = (const float*)d_in[1];
    const float* w_y  = (const float*)d_in[2];
    const float* w_x  = (const float*)d_in[3];
    const float* b_in = (const float*)d_in[4];
    const float* w_h  = (const float*)d_in[5];
    const float* b_h  = (const float*)d_in[6];
    const float* w_o  = (const float*)d_in[7];
    const float* b_o  = (const float*)d_in[8];
    const float* h0   = (const float*)d_in[9];
    float* out = (float*)d_out;

    char* ws = (char*)d_ws;
    __bf16* wyT = (__bf16*)ws;  ws += (size_t)UNITS * UNITS * sizeof(__bf16);
    __bf16* wxT = (__bf16*)ws;  ws += (size_t)UNITS * EMBED * sizeof(__bf16);
    __bf16* whT = (__bf16*)ws;  ws += (size_t)NLAYER * UNITS * UNITS * sizeof(__bf16);
    __bf16* woT = (__bf16*)ws;  ws += (size_t)UNITS * UNITS * sizeof(__bf16);
    __bf16* xeb = (__bf16*)ws;  ws += (size_t)SEQ * BATCH * EMBED * sizeof(__bf16);
    float*  P   = (float*)ws;   ws += (size_t)SEQ * BATCH * UNITS * sizeof(float);

    const int TPB = 256;
    auto blocks = [](size_t n, int tpb) { return (int)((n + tpb - 1) / tpb); };

    // weight transpose/convert to bf16
    tconv_kernel<<<blocks((size_t)UNITS * UNITS, TPB), TPB, 0, stream>>>(w_y, wyT, UNITS, UNITS);
    tconv_kernel<<<blocks((size_t)EMBED * UNITS, TPB), TPB, 0, stream>>>(w_x, wxT, EMBED, UNITS);
    tconv_kernel<<<blocks((size_t)UNITS * UNITS, TPB), TPB, 0, stream>>>(w_h, whT, UNITS, UNITS);
    tconv_kernel<<<blocks((size_t)UNITS * UNITS, TPB), TPB, 0, stream>>>(
        w_h + (size_t)UNITS * UNITS, whT + (size_t)UNITS * UNITS, UNITS, UNITS);
    tconv_kernel<<<blocks((size_t)UNITS * UNITS, TPB), TPB, 0, stream>>>(w_o, woT, UNITS, UNITS);

    // embedding gather -> time-major bf16
    gather_kernel<<<BATCH * SEQ, 128, 0, stream>>>(x, emb, xeb);

    // time-parallel input projection: P = xe @ Wx + b_in  (chip-wide WMMA GEMM)
    dim3 pg(SEQ * BATCH / 64, UNITS / 256);
    pregemm_kernel<<<pg, 512, 0, stream>>>(xeb, wxT, b_in, P);

    // persistent recurrent kernel: LDS = 33,280*2 + 66,048 = 132,608 B
    hw_rnn_kernel<<<BATCH / 16, 512, 132608, stream>>>(
        b_h, b_o, h0, wyT, whT, woT, P, out);
}
